// S5SSM_56057913148060
// MI455X (gfx1250) — matrix operand order repaired and checked by reading, not compiled, not run on hardware
//
#include <hip/hip_runtime.h>
#include <hip/hip_bf16.h>

// Problem constants (match reference)
#define LL 16384
#define HH 512
#define PP 512
#define TT 128            // scan chunk length
#define NC (LL / TT)      // 128 chunks
#define K2 (2 * PP)       // 1024, concatenated K for output GEMM

typedef __attribute__((ext_vector_type(16))) __bf16 v16bf;
typedef __attribute__((ext_vector_type(8)))  __bf16 v8bf;
typedef __attribute__((ext_vector_type(8)))  float  v8f;
typedef __attribute__((ext_vector_type(4)))  unsigned int v4u;
typedef __attribute__((ext_vector_type(8)))  int v8i;
typedef __attribute__((ext_vector_type(4)))  int v4i;

union U16 { v16bf v; v8bf h[2]; };

static __device__ __forceinline__ unsigned short f2bf(float f) {
    unsigned u = __float_as_uint(f);
    unsigned r = u + 0x7FFFu + ((u >> 16) & 1u);   // round-to-nearest-even
    return (unsigned short)(r >> 16);
}

// ---------------------------------------------------------------------------
// TDM: async 2-D tile load global -> LDS (D# built in SGPRs).
// data_size = 2 bytes. Optional LDS padding: +16B after every 1024B stored.
// ---------------------------------------------------------------------------
static __device__ __forceinline__ void tdm_load_2d_bf16(
    unsigned lds_addr, const void* gaddr,
    unsigned tile_w, unsigned tile_h,          // in elements / rows
    unsigned tensor_w, unsigned tensor_h,      // for OOB checks
    unsigned row_stride_elems, bool pad_en) {
    unsigned long long ga = (unsigned long long)gaddr;
    v4u g0;
    g0[0] = 1u;                                           // count=1, user D#
    g0[1] = lds_addr;                                     // LDS byte address
    g0[2] = (unsigned)(ga & 0xFFFFFFFFull);               // global addr lo
    g0[3] = (unsigned)((ga >> 32) & 0x1FFFFFFull) | (2u << 30);  // hi | type=2
    unsigned w0 = (1u << 16);                             // data_size=1 (2B)
    if (pad_en) w0 |= (1u << 20)                          // pad_enable
               | (7u << 22)                               // pad_interval: 256 dw = 1024B
               | (3u << 25);                              // pad_amount: 4 dw = 16B
    v8i g1;
    g1[0] = (int)w0;                                      // wg_mask=0, flags
    g1[1] = (int)((tensor_w & 0xFFFFu) << 16);            // tensor_dim0[15:0] @ bits63:48
    g1[2] = (int)(((tensor_w >> 16) & 0xFFFFu) | ((tensor_h & 0xFFFFu) << 16));
    g1[3] = (int)(((tensor_h >> 16) & 0xFFFFu) | ((tile_w & 0xFFFFu) << 16));
    g1[4] = (int)(tile_h & 0xFFFFu);                      // tile_dim1; tile_dim2=0
    g1[5] = (int)row_stride_elems;                        // tensor_dim0_stride lo32
    g1[6] = 0;
    g1[7] = 0;
    v4i z4 = {0, 0, 0, 0};
#if __clang_major__ >= 23
    v8i z8 = {0, 0, 0, 0, 0, 0, 0, 0};
    __builtin_amdgcn_tensor_load_to_lds(g0, g1, z4, z4, z8, 0);
#else
    __builtin_amdgcn_tensor_load_to_lds(g0, g1, z4, z4, 0);
#endif
}

// Same fragment but from LDS (two 16B ds loads at byte offsets o0, o1).
static __device__ __forceinline__ v16bf load_frag_a_lds(const char* s, unsigned o0,
                                                        unsigned o1) {
    U16 r;
    r.h[0] = *reinterpret_cast<const v8bf*>(s + o0);
    r.h[1] = *reinterpret_cast<const v8bf*>(s + o1);
    return r.v;
}
// B fragment (32x16 bf16, KxN): lane n=lane&15 holds column n (K 8h..8h+7 then 16+8h..23+8h, h=lane>>4).
static __device__ __forceinline__ v16bf load_frag_b(const unsigned short* p) {
    U16 r;
    r.h[0] = *reinterpret_cast<const v8bf*>(p);
    r.h[1] = *reinterpret_cast<const v8bf*>(p + 16);
    return r.v;
}

// ---------------------------------------------------------------------------
// Kernel 0: ZOH discretization. Bbar = ((exp(L*dt)-1)/L) * B_tilde, bf16 out.
// ---------------------------------------------------------------------------
__global__ __launch_bounds__(256) void k_disc(
    const float* __restrict__ Lre, const float* __restrict__ Lim,
    const float* __restrict__ Bre, const float* __restrict__ Bim,
    const float* __restrict__ logstep,
    unsigned short* __restrict__ BbR, unsigned short* __restrict__ BbI,
    float* __restrict__ lamR, float* __restrict__ lamI) {
    int idx = blockIdx.x * blockDim.x + threadIdx.x;
    if (idx >= PP * HH) return;
    int p = idx / HH, h = idx % HH;
    float st = __expf(logstep[p]);
    float lr = Lre[p], li = Lim[p];
    float er = __expf(lr * st);
    float s, c;
    __sincosf(li * st, &s, &c);
    float lbr = er * c, lbi = er * s;              // Lambda_bar
    float nr = lbr - 1.0f, ni = lbi;               // Lambda_bar - 1
    float den = lr * lr + li * li;
    float fr = (nr * lr + ni * li) / den;          // (Lbar-1)/Lambda
    float fi = (ni * lr - nr * li) / den;
    float br = Bre[idx], bi = Bim[idx];
    BbR[idx] = f2bf(fr * br - fi * bi);
    BbI[idx] = f2bf(fr * bi + fi * br);
    if (h == 0) { lamR[p] = lbr; lamI[p] = lbi; }
}

// ---------------------------------------------------------------------------
// Kernel 1: u (f32) -> bf16
// ---------------------------------------------------------------------------
__global__ __launch_bounds__(256) void k_u2bf(const float* __restrict__ u,
                                              unsigned short* __restrict__ uB) {
    int idx = blockIdx.x * blockDim.x + threadIdx.x;
    if (idx < LL * HH) uB[idx] = f2bf(u[idx]);
}

// ---------------------------------------------------------------------------
// Kernel 2: B2[n][k] = k<P ? C_re[n][k] : C_im[n][k-P]  (bf16, H x 1024)
// ---------------------------------------------------------------------------
__global__ __launch_bounds__(256) void k_b2(const float* __restrict__ Cre,
                                            const float* __restrict__ Cim,
                                            unsigned short* __restrict__ B2) {
    int idx = blockIdx.x * blockDim.x + threadIdx.x;
    if (idx >= HH * K2) return;
    int n = idx >> 10, k = idx & (K2 - 1);
    float v = (k < PP) ? Cre[n * PP + k] : Cim[n * PP + (k - PP)];
    B2[idx] = f2bf(v);
}

// ---------------------------------------------------------------------------
// Kernel 3: GEMM1  Bu_{re,im}[L,P] = u[L,H] @ Bbar_{re,im}^T   (bf16 WMMA)
// Block = 32-row stripe of u staged into LDS via TDM (padded rows, 1040B).
// 8 waves; each wave computes 2 M-tiles x (re,im) for 4 N-tiles.
// ---------------------------------------------------------------------------
__global__ __launch_bounds__(256) void k_gemm1(
    const unsigned short* __restrict__ uB,
    const unsigned short* __restrict__ BbR, const unsigned short* __restrict__ BbI,
    float* __restrict__ BuR, float* __restrict__ BuI) {
    __shared__ char sA[32 * 1040];                 // 32 rows x (1024B + 16B pad)
    const int mBase = blockIdx.x * 32;             // 2 M-tiles per block
    const int wave  = threadIdx.x >> 5;
    const int lane  = threadIdx.x & 31;

    if (threadIdx.x < 32) {
        // low 32 bits of the generic address of an LDS object = LDS byte offset
        unsigned ldsBase = (unsigned)(unsigned long long)(void*)sA;
        tdm_load_2d_bf16(ldsBase, uB + (size_t)mBase * HH,
                         /*tile_w=*/HH, /*tile_h=*/32,
                         /*tensor_w=*/HH, /*tensor_h=*/LL,
                         /*stride=*/HH, /*pad=*/true);
        __builtin_amdgcn_s_wait_tensorcnt(0);
    }
    // TDM wrote LDS behind the compiler's back: force sA to escape + clobber.
    asm volatile("" : : "r"(sA) : "memory");
    __syncthreads();

    const unsigned rA  = (unsigned)(lane & 15);    // fragment row within tile
    const unsigned kab = 16u * (unsigned)(lane >> 4); // ka bytes
    const unsigned rowB = (unsigned)(lane & 15);
    const unsigned kbb  = 8u * (unsigned)(lane >> 4); // kb elements

    for (int j = 0; j < 4; ++j) {
        int tn = j * 8 + wave;                     // N tile (0..31)
        const unsigned short* bPr = BbR + (size_t)(tn * 16 + rowB) * HH + kbb;
        const unsigned short* bPi = BbI + (size_t)(tn * 16 + rowB) * HH + kbb;
        v8f acc0R = {}, acc0I = {}, acc1R = {}, acc1I = {};
#pragma unroll
        for (int k = 0; k < HH; k += 32) {
            unsigned o = (unsigned)(k * 2) + kab;  // byte offset in 1024B row
            v16bf a0 = load_frag_a_lds(sA, rA * 1040u + o, rA * 1040u + o + 32u);
            v16bf a1 = load_frag_a_lds(sA, (rA + 16u) * 1040u + o,
                                           (rA + 16u) * 1040u + o + 32u);
            v16bf br = load_frag_b(bPr + k);
            v16bf bi = load_frag_b(bPi + k);
            acc0R = __builtin_amdgcn_wmma_f32_16x16x32_bf16(false, a0, false, br,
                                                            (short)0, acc0R, false, false);
            acc0I = __builtin_amdgcn_wmma_f32_16x16x32_bf16(false, a0, false, bi,
                                                            (short)0, acc0I, false, false);
            acc1R = __builtin_amdgcn_wmma_f32_16x16x32_bf16(false, a1, false, br,
                                                            (short)0, acc1R, false, false);
            acc1I = __builtin_amdgcn_wmma_f32_16x16x32_bf16(false, a1, false, bi,
                                                            (short)0, acc1I, false, false);
        }
        int col   = tn * 16 + (lane & 15);
        int rbase = mBase + 8 * (lane >> 4);
        for (int v = 0; v < 8; ++v) {
            BuR[(size_t)(rbase + v) * PP + col]      = acc0R[v];
            BuI[(size_t)(rbase + v) * PP + col]      = acc0I[v];
            BuR[(size_t)(rbase + 16 + v) * PP + col] = acc1R[v];
            BuI[(size_t)(rbase + 16 + v) * PP + col] = acc1I[v];
        }
    }
}

// ---------------------------------------------------------------------------
// Kernel 4: local chunk scan (in-place on Bu), x_t = lam*x_{t-1} + Bu_t.
// ---------------------------------------------------------------------------
__global__ __launch_bounds__(PP) void k_scan_local(
    float* __restrict__ BuR, float* __restrict__ BuI,
    const float* __restrict__ lamR, const float* __restrict__ lamI,
    float* __restrict__ ER, float* __restrict__ EI) {
    int p = threadIdx.x;
    int c = blockIdx.x;
    float lr = lamR[p], li = lamI[p];
    float xr = 0.f, xi = 0.f;
    int base = c * TT * PP + p;
    for (int i = 0; i < TT; ++i) {
        float br = BuR[base + i * PP];
        float bi = BuI[base + i * PP];
        float nr = lr * xr - li * xi + br;
        float ni = lr * xi + li * xr + bi;
        xr = nr; xi = ni;
        BuR[base + i * PP] = xr;
        BuI[base + i * PP] = xi;
    }
    ER[c * PP + p] = xr;
    EI[c * PP + p] = xi;
}

// ---------------------------------------------------------------------------
// Kernel 5: carry scan across chunks.
// ---------------------------------------------------------------------------
__global__ __launch_bounds__(PP) void k_scan_carry(
    const float* __restrict__ lamR, const float* __restrict__ lamI,
    const float* __restrict__ ER, const float* __restrict__ EI,
    float* __restrict__ CR, float* __restrict__ CI) {
    int p = threadIdx.x;
    float lr = lamR[p], li = lamI[p];
    float ar = lr, ai = li;                        // lam^128 by 7 squarings
    for (int s = 0; s < 7; ++s) {
        float nr = ar * ar - ai * ai;
        float ni = 2.f * ar * ai;
        ar = nr; ai = ni;
    }
    float cr = 0.f, ci = 0.f;
    for (int c = 0; c < NC; ++c) {
        CR[c * PP + p] = cr;
        CI[c * PP + p] = ci;
        float er = ER[c * PP + p], ei = EI[c * PP + p];
        float nr = ar * cr - ai * ci + er;
        float ni = ar * ci + ai * cr + ei;
        cr = nr; ci = ni;
    }
}

// ---------------------------------------------------------------------------
// Kernel 6: fixup x_t += lam^{i+1}*carry, pack A2 = [x_re | -x_im] as bf16.
// ---------------------------------------------------------------------------
__global__ __launch_bounds__(PP) void k_fixup(
    const float* __restrict__ BuR, const float* __restrict__ BuI,
    const float* __restrict__ lamR, const float* __restrict__ lamI,
    const float* __restrict__ CR, const float* __restrict__ CI,
    unsigned short* __restrict__ A2) {
    int p = threadIdx.x;
    int c = blockIdx.x;
    float lr = lamR[p], li = lamI[p];
    float cr = CR[c * PP + p], ci = CI[c * PP + p];
    float pwr = lr, pwi = li;                       // lam^{i+1}
    for (int i = 0; i < TT; ++i) {
        int t = c * TT + i;
        float xr = BuR[t * PP + p] + (pwr * cr - pwi * ci);
        float xi = BuI[t * PP + p] + (pwr * ci + pwi * cr);
        float nr = pwr * lr - pwi * li;
        float ni = pwr * li + pwi * lr;
        pwr = nr; pwi = ni;
        A2[(size_t)t * K2 + p]      = f2bf(xr);
        A2[(size_t)t * K2 + PP + p] = f2bf(-xi);    // fold the Re(...) minus sign
    }
}

// ---------------------------------------------------------------------------
// Kernel 7: GEMM2  ys[L,H] = A2[L,1024] @ B2^T + D*u   (bf16 WMMA, f32 out)
// Block = 32-row stripe of A2 staged via TDM (rows 2048B + 2x16B pads = 2080B).
// ---------------------------------------------------------------------------
__global__ __launch_bounds__(256) void k_gemm2(
    const unsigned short* __restrict__ A2, const unsigned short* __restrict__ B2,
    const float* __restrict__ u, const float* __restrict__ Dv,
    float* __restrict__ out) {
    __shared__ char sA[32 * 2080];                 // 66560 B
    const int mBase = blockIdx.x * 32;
    const int wave  = threadIdx.x >> 5;
    const int lane  = threadIdx.x & 31;

    if (threadIdx.x < 32) {
        unsigned ldsBase = (unsigned)(unsigned long long)(void*)sA;
        tdm_load_2d_bf16(ldsBase, A2 + (size_t)mBase * K2,
                         /*tile_w=*/K2, /*tile_h=*/32,
                         /*tensor_w=*/K2, /*tensor_h=*/LL,
                         /*stride=*/K2, /*pad=*/true);
        __builtin_amdgcn_s_wait_tensorcnt(0);
    }
    asm volatile("" : : "r"(sA) : "memory");
    __syncthreads();

    const unsigned rA  = (unsigned)(lane & 15);
    const unsigned kab = 16u * (unsigned)(lane >> 4);
    const unsigned rowB = (unsigned)(lane & 15);
    const unsigned kbb  = 8u * (unsigned)(lane >> 4);

    for (int j = 0; j < 4; ++j) {
        int tn = j * 8 + wave;                     // N tile (0..31)
        const unsigned short* bP = B2 + (size_t)(tn * 16 + rowB) * K2 + kbb;
        v8f acc0 = {}, acc1 = {};
#pragma unroll
        for (int k = 0; k < K2; k += 32) {
            unsigned o  = (unsigned)(k * 2) + kab;            // 0..2030
            unsigned p0 = o  + ((o  >= 1024u) ? 16u : 0u);    // pad-adjusted
            unsigned o2 = o + 32u;
            unsigned p1 = o2 + ((o2 >= 1024u) ? 16u : 0u);
            v16bf a0 = load_frag_a_lds(sA, rA * 2080u + p0, rA * 2080u + p1);
            v16bf a1 = load_frag_a_lds(sA, (rA + 16u) * 2080u + p0,
                                           (rA + 16u) * 2080u + p1);
            v16bf b  = load_frag_b(bP + k);
            acc0 = __builtin_amdgcn_wmma_f32_16x16x32_bf16(false, a0, false, b,
                                                           (short)0, acc0, false, false);
            acc1 = __builtin_amdgcn_wmma_f32_16x16x32_bf16(false, a1, false, b,
                                                           (short)0, acc1, false, false);
        }
        int col   = tn * 16 + (lane & 15);
        int rbase = mBase + 8 * (lane >> 4);
        float dv  = Dv[col];
        for (int v = 0; v < 8; ++v) {
            int r0 = rbase + v, r1 = rbase + 16 + v;
            out[(size_t)r0 * HH + col] = acc0[v] + dv * u[(size_t)r0 * HH + col];
            out[(size_t)r1 * HH + col] = acc1[v] + dv * u[(size_t)r1 * HH + col];
        }
    }
}

// ---------------------------------------------------------------------------
extern "C" void kernel_launch(void* const* d_in, const int* in_sizes, int n_in,
                              void* d_out, int out_size, void* d_ws, size_t ws_size,
                              hipStream_t stream) {
    const float* u       = (const float*)d_in[0];
    const float* Lre     = (const float*)d_in[1];
    const float* Lim     = (const float*)d_in[2];
    const float* Bre     = (const float*)d_in[3];
    const float* Bim     = (const float*)d_in[4];
    const float* Cre     = (const float*)d_in[5];
    const float* Cim     = (const float*)d_in[6];
    const float* Dv      = (const float*)d_in[7];
    const float* logstep = (const float*)d_in[8];
    float* out = (float*)d_out;

    // Workspace layout (bytes, all 16B aligned)
    char* w = (char*)d_ws;
    unsigned short* uB  = (unsigned short*)(w);                          // L*H bf16
    unsigned short* BbR = (unsigned short*)(w + 16777216ull);            // P*H bf16
    unsigned short* BbI = (unsigned short*)(w + 17301504ull);            // P*H bf16
    float* BuR  = (float*)(w + 17825792ull);                             // L*P f32
    float* BuI  = (float*)(w + 51380224ull);                             // L*P f32
    float* lamR = (float*)(w + 84934656ull);                             // P f32
    float* lamI = (float*)(w + 84936704ull);                             // P f32
    float* ER   = (float*)(w + 84938752ull);                             // NC*P f32
    float* EI   = (float*)(w + 85200896ull);                             // NC*P f32
    float* CR   = (float*)(w + 85463040ull);                             // NC*P f32
    float* CI   = (float*)(w + 85725184ull);                             // NC*P f32
    unsigned short* A2 = (unsigned short*)(w + 85987328ull);             // L*1024 bf16
    unsigned short* B2 = (unsigned short*)(w + 119541760ull);            // H*1024 bf16

    k_disc<<<(PP * HH) / 256, 256, 0, stream>>>(Lre, Lim, Bre, Bim, logstep,
                                                BbR, BbI, lamR, lamI);
    k_u2bf<<<(LL * HH) / 256, 256, 0, stream>>>(u, uB);
    k_b2<<<(HH * K2) / 256, 256, 0, stream>>>(Cre, Cim, B2);

    // GEMM1: 512 blocks, each owns a 32-row stripe staged via TDM
    k_gemm1<<<LL / 32, 256, 0, stream>>>(uB, BbR, BbI, BuR, BuI);

    k_scan_local<<<NC, PP, 0, stream>>>(BuR, BuI, lamR, lamI, ER, EI);
    k_scan_carry<<<1, PP, 0, stream>>>(lamR, lamI, ER, EI, CR, CI);
    k_fixup<<<NC, PP, 0, stream>>>(BuR, BuI, lamR, lamI, CR, CI, A2);

    // GEMM2: 512 blocks
    k_gemm2<<<LL / 32, 256, 0, stream>>>(A2, B2, u, Dv, out);
}
